// MultiHeadAttention_66391604461983
// MI455X (gfx1250) — matrix operand order repaired and checked by reading, not compiled, not run on hardware
//
#include <hip/hip_runtime.h>
#include <hip/hip_bf16.h>

typedef __attribute__((ext_vector_type(16))) __bf16 v16bf;
typedef __attribute__((ext_vector_type(8)))  float  v8f;

#define E_DIM 1024
#define S_DIM 2048
#define H_NUM 16
#define DH_DIM 64
#define B_NUM 2

__device__ __forceinline__ unsigned short f2bf(float f) {
    unsigned int u = __builtin_bit_cast(unsigned int, f);
    u += 0x7FFFu + ((u >> 16) & 1u);        // round-to-nearest-even
    return (unsigned short)(u >> 16);
}

union Frag { v16bf v; uint4 q[2]; unsigned short u[16]; };
union Acc  { v8f v; float f[8]; };

// Build a 16-element bf16 fragment from two 16-byte contiguous LDS runs.
__device__ __forceinline__ v16bf ld_frag(const unsigned short* p0,
                                         const unsigned short* p1) {
    Frag fr;
    fr.q[0] = *(const uint4*)p0;
    fr.q[1] = *(const uint4*)p1;
    return fr.v;
}

// ---------------------------------------------------------------------------
// Kernel 1: fused QKV projection GEMM (M=4096, N=1024, K=1024), f32 in,
// bf16 out in [B,H,S,DH] layout using the reference's no-transpose reshape:
//   dst[b,h,s,d] = (X W + b)[b, s2 = h*128 + s/16, e = (s%16)*64 + d]
// Tile: BM=128, BN=128, BK=32; 8 waves, each wave computes 32x64 (2x4 WMMA).
// ---------------------------------------------------------------------------
__global__ __launch_bounds__(256)
void qkv_proj_kernel(const float* __restrict__ xq, const float* __restrict__ xk,
                     const float* __restrict__ xv,
                     const float* __restrict__ Wq, const float* __restrict__ Wk,
                     const float* __restrict__ Wv,
                     const float* __restrict__ bq, const float* __restrict__ bk,
                     const float* __restrict__ bv,
                     unsigned short* __restrict__ Qh,
                     unsigned short* __restrict__ Kh,
                     unsigned short* __restrict__ Vh)
{
    __shared__ unsigned short sA[128 * 40];   // [m][k] bf16, pad to 40 (16B aligned rows)
    __shared__ unsigned short sB[128 * 40];   // [n][k] bf16 (W tile transposed)

    const int tid  = threadIdx.x;
    const int lane = tid & 31;
    const int wid  = tid >> 5;
    const int wm   = wid >> 1;                // 0..3 -> 32-row slice
    const int wn   = wid & 1;                 // 0..1 -> 64-col slice
    const int half = lane >> 4;
    const int l16  = lane & 15;

    const float* X; const float* W; const float* bias; unsigned short* dst;
    if (blockIdx.z == 0)      { X = xq; W = Wq; bias = bq; dst = Qh; }
    else if (blockIdx.z == 1) { X = xk; W = Wk; bias = bk; dst = Kh; }
    else                      { X = xv; W = Wv; bias = bv; dst = Vh; }

    const int rowBase = blockIdx.y * 128;     // over B*S = 4096 rows
    const int colBase = blockIdx.x * 128;     // over E = 1024 cols

    Acc acc[2][4];
    #pragma unroll
    for (int i = 0; i < 2; ++i)
        #pragma unroll
        for (int j = 0; j < 4; ++j)
            #pragma unroll
            for (int r = 0; r < 8; ++r) acc[i][j].f[r] = 0.0f;

    for (int kt = 0; kt < E_DIM / 32; ++kt) {
        // ---- stage A tile (128 x 32 f32 -> bf16) ----
        #pragma unroll
        for (int j = 0; j < 4; ++j) {
            int id = tid + j * 256;
            int r  = id >> 3;
            int c4 = (id & 7) * 4;
            float4 f = *(const float4*)&X[(size_t)(rowBase + r) * E_DIM + kt * 32 + c4];
            unsigned short* p = &sA[r * 40 + c4];
            p[0] = f2bf(f.x); p[1] = f2bf(f.y); p[2] = f2bf(f.z); p[3] = f2bf(f.w);
        }
        // ---- stage B tile transposed (W 32 x 128 f32 -> sB[n][k] bf16) ----
        #pragma unroll
        for (int j = 0; j < 4; ++j) {
            int id = tid + j * 256;
            int k  = id >> 5;
            int n4 = (id & 31) * 4;
            float4 f = *(const float4*)&W[(size_t)(kt * 32 + k) * E_DIM + colBase + n4];
            sB[(n4 + 0) * 40 + k] = f2bf(f.x);
            sB[(n4 + 1) * 40 + k] = f2bf(f.y);
            sB[(n4 + 2) * 40 + k] = f2bf(f.z);
            sB[(n4 + 3) * 40 + k] = f2bf(f.w);
        }
        // prefetch next K-tile into cache hierarchy
        if (kt + 1 < E_DIM / 32) {
            __builtin_prefetch(&X[(size_t)(rowBase + (tid >> 3)) * E_DIM + (kt + 1) * 32], 0, 1);
            __builtin_prefetch(&W[(size_t)((kt + 1) * 32 + (tid >> 5)) * E_DIM + colBase], 0, 1);
        }
        __syncthreads();

        // ---- fragment gathers (ISA 7.12.2 bf16 layouts) ----
        v16bf afr[2], bfr[4];
        #pragma unroll
        for (int tm = 0; tm < 2; ++tm) {
            int m = wm * 32 + tm * 16 + l16;
            afr[tm] = ld_frag(&sA[m * 40 + half * 8], &sA[m * 40 + 16 + half * 8]);
        }
        #pragma unroll
        for (int tn = 0; tn < 4; ++tn) {
            int n = wn * 64 + tn * 16 + l16;
            bfr[tn] = ld_frag(&sB[n * 40 + half * 8], &sB[n * 40 + 16 + half * 8]);
        }
        #pragma unroll
        for (int tm = 0; tm < 2; ++tm)
            #pragma unroll
            for (int tn = 0; tn < 4; ++tn)
                acc[tm][tn].v = __builtin_amdgcn_wmma_f32_16x16x32_bf16(
                    false, afr[tm], false, bfr[tn], (short)0, acc[tm][tn].v,
                    false, false);
        __syncthreads();
    }

    // ---- epilogue: bias add, bf16 convert, bug-faithful head remap ----
    #pragma unroll
    for (int tm = 0; tm < 2; ++tm) {
        #pragma unroll
        for (int tn = 0; tn < 4; ++tn) {
            int col = colBase + wn * 64 + tn * 16 + l16;
            float bv_ = bias[col];
            #pragma unroll
            for (int r = 0; r < 8; ++r) {
                int row = rowBase + wm * 32 + tm * 16 + half * 8 + r;
                float val = acc[tm][tn].f[r] + bv_;
                int b  = row >> 11;
                int s2 = row & (S_DIM - 1);
                int hh = s2 >> 7;
                int s  = ((s2 & 127) << 4) + (col >> 6);
                int d  = col & 63;
                dst[(((size_t)(b * H_NUM + hh) * S_DIM) + s) * DH_DIM + d] = f2bf(val);
            }
        }
    }
}

// ---------------------------------------------------------------------------
// Kernel 2: causal flash attention over bf16 [B,H,S,DH] with f32 softmax.
// Block = 8 waves, 128 queries; wave owns 16 queries; key tiles of 32.
// ---------------------------------------------------------------------------
__global__ __launch_bounds__(256)
void attention_kernel(const unsigned short* __restrict__ Qh,
                      const unsigned short* __restrict__ Kh,
                      const unsigned short* __restrict__ Vh,
                      float* __restrict__ out)
{
    __shared__ unsigned short sQ[128 * 72];     // [q][dh]
    __shared__ unsigned short sK[32 * 72];      // [key][dh]
    __shared__ unsigned short sVt[64 * 40];     // [dh][key] (transposed)
    __shared__ unsigned short sP[8 * 16 * 40];  // per-wave P scratch [m][key]

    const int tid  = threadIdx.x;
    const int lane = tid & 31;
    const int wid  = tid >> 5;
    const int half = lane >> 4;
    const int l16  = lane & 15;
    const int b = blockIdx.z, h = blockIdx.y;
    const int qBlock = blockIdx.x * 128;
    const size_t headOff = (size_t)(b * H_NUM + h) * S_DIM * DH_DIM;

    // stage Q tile once
    #pragma unroll
    for (int j = 0; j < 4; ++j) {
        int id = tid + j * 256;
        int r  = id >> 3;
        int c8 = (id & 7) * 8;
        uint4 v = *(const uint4*)&Qh[headOff + (size_t)(qBlock + r) * DH_DIM + c8];
        *(uint4*)&sQ[r * 72 + c8] = v;
    }
    __syncthreads();

    const int qWave = qBlock + wid * 16;        // wave's first query row
    const int qm = wid * 16 + l16;
    v16bf qa0 = ld_frag(&sQ[qm * 72 + half * 8],      &sQ[qm * 72 + 16 + half * 8]);
    v16bf qa1 = ld_frag(&sQ[qm * 72 + 32 + half * 8], &sQ[qm * 72 + 48 + half * 8]);

    Acc o[4];
    #pragma unroll
    for (int t = 0; t < 4; ++t)
        #pragma unroll
        for (int r = 0; r < 8; ++r) o[t].f[r] = 0.0f;
    float mrow[8], lrow[8];
    #pragma unroll
    for (int r = 0; r < 8; ++r) { mrow[r] = -1.0e30f; lrow[r] = 0.0f; }

    unsigned short* sPw = &sP[wid * 16 * 40];
    const int kbEnd = qBlock + 128;             // causal frontier for this block

    for (int kb = 0; kb < kbEnd; kb += 32) {
        // ---- stage K tile + transposed V tile ----
        {
            int r  = tid >> 3;
            int c8 = (tid & 7) * 8;
            uint4 kv = *(const uint4*)&Kh[headOff + (size_t)(kb + r) * DH_DIM + c8];
            *(uint4*)&sK[r * 72 + c8] = kv;
            uint4 vv = *(const uint4*)&Vh[headOff + (size_t)(kb + r) * DH_DIM + c8];
            const unsigned short* vs = (const unsigned short*)&vv;
            #pragma unroll
            for (int jj = 0; jj < 8; ++jj)
                sVt[(c8 + jj) * 40 + r] = vs[jj];
        }
        __syncthreads();

        if (qWave + 15 >= kb) {                 // wave has at least one live key
            // ---- scores S = Q K^T (two 16-key tiles, K-depth 64 = 2 wmma) ----
            Acc sc[2];
            #pragma unroll
            for (int nt = 0; nt < 2; ++nt) {
                int key = nt * 16 + l16;
                v16bf kb0 = ld_frag(&sK[key * 72 + half * 8],
                                    &sK[key * 72 + 16 + half * 8]);
                v16bf kb1 = ld_frag(&sK[key * 72 + 32 + half * 8],
                                    &sK[key * 72 + 48 + half * 8]);
                Acc z;
                #pragma unroll
                for (int r = 0; r < 8; ++r) z.f[r] = 0.0f;
                z.v = __builtin_amdgcn_wmma_f32_16x16x32_bf16(
                    false, qa0, false, kb0, (short)0, z.v, false, false);
                z.v = __builtin_amdgcn_wmma_f32_16x16x32_bf16(
                    false, qa1, false, kb1, (short)0, z.v, false, false);
                sc[nt] = z;
            }
            // ---- scale + causal mask (analytic; 1/sqrt(64) = 0.125) ----
            #pragma unroll
            for (int nt = 0; nt < 2; ++nt) {
                int key = kb + nt * 16 + l16;
                #pragma unroll
                for (int r = 0; r < 8; ++r) {
                    int qrow = qWave + half * 8 + r;
                    float v = sc[nt].f[r] * 0.125f;
                    sc[nt].f[r] = (key > qrow) ? -1.0e30f : v;
                }
            }
            // ---- online softmax (row reductions across 16-lane halves) ----
            float alpha[8];
            #pragma unroll
            for (int r = 0; r < 8; ++r) {
                float mx = fmaxf(sc[0].f[r], sc[1].f[r]);
                #pragma unroll
                for (int off = 8; off > 0; off >>= 1)
                    mx = fmaxf(mx, __shfl_xor(mx, off, 32));
                float mn = fmaxf(mrow[r], mx);
                alpha[r] = __expf(mrow[r] - mn);
                mrow[r]  = mn;
                float p0 = __expf(sc[0].f[r] - mn);
                float p1 = __expf(sc[1].f[r] - mn);
                sc[0].f[r] = p0; sc[1].f[r] = p1;
                float ls = p0 + p1;
                #pragma unroll
                for (int off = 8; off > 0; off >>= 1)
                    ls += __shfl_xor(ls, off, 32);
                lrow[r] = lrow[r] * alpha[r] + ls;
            }
            #pragma unroll
            for (int t = 0; t < 4; ++t)
                #pragma unroll
                for (int r = 0; r < 8; ++r) o[t].f[r] *= alpha[r];

            // ---- P: C-layout -> A-layout via per-wave LDS round trip ----
            #pragma unroll
            for (int nt = 0; nt < 2; ++nt)
                #pragma unroll
                for (int r = 0; r < 8; ++r)
                    sPw[(half * 8 + r) * 40 + nt * 16 + l16] = f2bf(sc[nt].f[r]);
            asm volatile("s_wait_dscnt 0" ::: "memory");  // per-wave LDS fence
            v16bf pa = ld_frag(&sPw[l16 * 40 + half * 8],
                               &sPw[l16 * 40 + 16 + half * 8]);

            // ---- O += P V (4 dh tiles) ----
            #pragma unroll
            for (int t = 0; t < 4; ++t) {
                int dh = t * 16 + l16;
                v16bf vb = ld_frag(&sVt[dh * 40 + half * 8],
                                   &sVt[dh * 40 + 16 + half * 8]);
                o[t].v = __builtin_amdgcn_wmma_f32_16x16x32_bf16(
                    false, pa, false, vb, (short)0, o[t].v, false, false);
            }
        }
        __syncthreads();
    }

    // ---- normalize + bug-faithful output reshape store (f32) ----
    #pragma unroll
    for (int t = 0; t < 4; ++t) {
        #pragma unroll
        for (int r = 0; r < 8; ++r) {
            int qrow = qWave + half * 8 + r;
            int d    = t * 16 + l16;
            float val = o[t].f[r] / lrow[r];
            int s2 = h * 128 + (qrow >> 4);
            int e  = ((qrow & 15) << 6) + d;
            out[((size_t)b * S_DIM + s2) * E_DIM + e] = val;
        }
    }
}

extern "C" void kernel_launch(void* const* d_in, const int* in_sizes, int n_in,
                              void* d_out, int out_size, void* d_ws, size_t ws_size,
                              hipStream_t stream) {
    (void)in_sizes; (void)n_in; (void)out_size; (void)ws_size;
    const float* q  = (const float*)d_in[0];
    const float* k  = (const float*)d_in[1];
    const float* v  = (const float*)d_in[2];
    const float* Wq = (const float*)d_in[3];
    const float* bq = (const float*)d_in[4];
    const float* Wk = (const float*)d_in[5];
    const float* bk = (const float*)d_in[6];
    const float* Wv = (const float*)d_in[7];
    const float* bv = (const float*)d_in[8];
    // d_in[9] = mask: causal, computed analytically in-kernel.

    unsigned short* Qh = (unsigned short*)d_ws;
    unsigned short* Kh = Qh + (size_t)B_NUM * S_DIM * E_DIM;
    unsigned short* Vh = Kh + (size_t)B_NUM * S_DIM * E_DIM;
    float* out = (float*)d_out;

    qkv_proj_kernel<<<dim3(E_DIM / 128, (B_NUM * S_DIM) / 128, 3), 256, 0, stream>>>(
        q, k, v, Wq, Wk, Wv, bq, bk, bv, Qh, Kh, Vh);
    attention_kernel<<<dim3(S_DIM / 128, H_NUM, B_NUM), 256, 0, stream>>>(
        Qh, Kh, Vh, out);
}
